// SingleHeadAttention_40742059770318
// MI455X (gfx1250) — matrix operand order repaired and checked by reading, not compiled, not run on hardware
//
#include <hip/hip_runtime.h>
#include <hip/hip_bf16.h>
#include <stdint.h>

// ---------------------------------------------------------------------------
// Single-head causal attention, MI455X (gfx1250, wave32, WMMA).
// cast(f32->f16) -> QKV GEMM (WMMA, async-LDS double-buffered B tiles)
//   -> causal scores GEMM (WMMA, 4 tiles/wave) -> row softmax (f32, in place)
//   -> P@V GEMM (WMMA, 4 tiles/wave, causal-bounded K loop) -> f32 out.
// GEMMs: v_wmma_f32_16x16x32_f16, A/B fragments use contiguous K order (k = 16*(lane/16) + s).
// Async staging: global_load_async_to_lds_b128 + s_wait_asynccnt (ISA 10/15.18)
// via inline asm (portable across ROCm7.2 / amdgpu-toolchain per bridge doc).
// ---------------------------------------------------------------------------

typedef _Float16 f16;
typedef __attribute__((ext_vector_type(16))) _Float16 v16h;
typedef __attribute__((ext_vector_type(8)))  _Float16 v8h;
typedef __attribute__((ext_vector_type(4)))  _Float16 v4h;
typedef __attribute__((ext_vector_type(8)))  float    v8f;

#define ATT_B 8
#define ATT_N 2048
#define ATT_D 1024
#define ATT_M (ATT_B * ATT_N)            // 16384 tokens total
#define NEG_INF (-__builtin_huge_valf())

static __device__ __forceinline__ v16h make_afrag(const f16* p) {
  // A-fragment (16x32 f16, MxK): lane&15 = row M, lane>>4 selects K sub-range.
  // Per lane: 16 contiguous f16 at +0 and +8 (pre-offset by halfsel*16).
  v8h lo = *(const v8h*)(p);
  v8h hi = *(const v8h*)(p + 8);
  return __builtin_shufflevector(lo, hi, 0, 1, 2, 3, 4, 5, 6, 7,
                                         8, 9, 10, 11, 12, 13, 14, 15);
}

static __device__ __forceinline__ v8f wmma_f16(v16h a, v16h b, v8f c) {
  return __builtin_amdgcn_wmma_f32_16x16x32_f16(
      /*neg_a=*/false, a, /*neg_b=*/false, b,
      /*c_mod=*/(short)0, c, /*reuse_a=*/false, /*reuse_b=*/false);
}

// Async copy of 16 bytes/lane from global to LDS (one VMEM instr per wave,
// ASYNCcnt += 1).  VDST VGPR carries the wave-relative LDS byte offset
// (low 32 bits of the flat shared-aperture address).
static __device__ __forceinline__ void stage_async_b128(const f16* g, f16* lds) {
  asm volatile("global_load_async_to_lds_b128 %0, %1, off"
               :: "v"((uint32_t)(uintptr_t)lds), "v"(g)
               : "memory");
}
static __device__ __forceinline__ void wait_async_le1() {
  asm volatile("s_wait_asynccnt 0x1" ::: "memory");
}
static __device__ __forceinline__ void wait_async_le0() {
  asm volatile("s_wait_asynccnt 0x0" ::: "memory");
}

// ---------------------------------------------------------------------------
// Kernel 1: f32 -> f16 cast, vectorized (b128 load, b64 store). n % 1024 == 0.
// ---------------------------------------------------------------------------
__global__ void k_cast_f32_f16(const float* __restrict__ src,
                               f16* __restrict__ dst, int n4) {
  int i = blockIdx.x * blockDim.x + threadIdx.x;
  if (i < n4) {
    float4 v = ((const float4*)src)[i];
    v4h o = { (f16)v.x, (f16)v.y, (f16)v.z, (f16)v.w };
    ((v4h*)dst)[i] = o;
  }
}

// ---------------------------------------------------------------------------
// Kernel 2: Q/K/V projection.  Y = Xh @ W^T; W is [out,in] so B[k][n] = W[n][k]
// -> the 64x32 B tile for the whole block is 64 contiguous-row chunks of W.
// All 8 waves share it: staged once per K-step into LDS via async copy,
// double-buffered (software pipeline: issue step i+2 while computing step i).
// Each wave: 16 M-rows x 64 N-cols, 4 WMMAs per K-step.  z==2 stores V^T.
// ---------------------------------------------------------------------------
#define BROWS   64
#define BSTRIDE 40   // f16 units; 80B row stride = 20 banks -> conflict-free frags

__global__ void k_qkv_gemm(const f16* __restrict__ Xh,
                           const f16* __restrict__ Wh,
                           f16* __restrict__ QK,      // Q then K, contiguous
                           f16* __restrict__ Vt) {    // [B, D, N]
  __shared__ f16 Bs[2][BROWS * BSTRIDE];

  const int tid     = threadIdx.x;
  const int lane    = tid & 31;
  const int wave    = tid >> 5;
  const int halfsel = lane >> 4;
  const int lr      = lane & 15;
  const int z       = blockIdx.z;                       // 0=Q 1=K 2=V
  const int m0      = (blockIdx.x * 8 + wave) * 16;
  const int n0      = blockIdx.y * 64;

  const f16* W    = Wh + (size_t)z * ATT_D * ATT_D;
  const f16* arow = Xh + (size_t)(m0 + lr) * ATT_D + halfsel * 16;

  // staging assignment: thread t copies 16B: row = t/4, chunk = t%4
  const int  srow   = tid >> 2;
  const int  schunk = (tid & 3) * 8;
  const f16* gsrc   = W + (size_t)(n0 + srow) * ATT_D + schunk;
  f16* ldst[2] = { &Bs[0][srow * BSTRIDE + schunk],
                   &Bs[1][srow * BSTRIDE + schunk] };

  stage_async_b128(gsrc + 0,  ldst[0]);   // K-step 0
  stage_async_b128(gsrc + 32, ldst[1]);   // K-step 1

  v8f acc0 = {}, acc1 = {}, acc2 = {}, acc3 = {};
  v16h a_cur = make_afrag(arow);

  for (int i = 0; i < 32; ++i) {
    v16h a_next = (i + 1 < 32) ? make_afrag(arow + (i + 1) * 32) : a_cur;

    if (i < 31) wait_async_le1(); else wait_async_le0();  // copy i complete
    __syncthreads();                                      // ... in all waves

    const f16* bb = &Bs[i & 1][halfsel * 16];
    acc0 = wmma_f16(a_cur, *(const v16h*)(bb + (lr +  0) * BSTRIDE), acc0);
    acc1 = wmma_f16(a_cur, *(const v16h*)(bb + (lr + 16) * BSTRIDE), acc1);
    acc2 = wmma_f16(a_cur, *(const v16h*)(bb + (lr + 32) * BSTRIDE), acc2);
    acc3 = wmma_f16(a_cur, *(const v16h*)(bb + (lr + 48) * BSTRIDE), acc3);

    __syncthreads();                                      // all done reading
    if (i + 2 < 32) stage_async_b128(gsrc + (i + 2) * 32, ldst[i & 1]);
    a_cur = a_next;
  }

  if (z != 2) {
    f16* out = QK + (size_t)z * ATT_M * ATT_D;
    for (int r = 0; r < 8; ++r) {
      size_t row = (size_t)(m0 + halfsel * 8 + r) * ATT_D;
      out[row + n0 + 0  + lr] = (f16)acc0[r];
      out[row + n0 + 16 + lr] = (f16)acc1[r];
      out[row + n0 + 32 + lr] = (f16)acc2[r];
      out[row + n0 + 48 + lr] = (f16)acc3[r];
    }
  } else {
    // V transposed: Vt[b*D*N + d*N + token]; tiles never straddle batches.
    int b  = m0 >> 11;
    int nn = (m0 & 2047) + halfsel * 8;
    size_t base = (size_t)b * ATT_D * ATT_N;
    for (int r = 0; r < 8; ++r) {
      Vt[base + (size_t)(n0 + 0  + lr) * ATT_N + nn + r] = (f16)acc0[r];
      Vt[base + (size_t)(n0 + 16 + lr) * ATT_N + nn + r] = (f16)acc1[r];
      Vt[base + (size_t)(n0 + 32 + lr) * ATT_N + nn + r] = (f16)acc2[r];
      Vt[base + (size_t)(n0 + 48 + lr) * ATT_N + nn + r] = (f16)acc3[r];
    }
  }
}

// ---------------------------------------------------------------------------
// Kernel 3: S = (Q @ K^T)/32, causal, f16 store.  Each wave: one 16-row query
// tile x four 16-col key tiles (nt group), reusing the A fragment 4x.
// Rectangular (qt x 32 nt-group) dispatch; waves with an empty group exit
// uniformly.  Groups straddling the diagonal compute all 4 tiles but store
// only nt <= qt (<=2.3% wasted WMMAs overall).
// ---------------------------------------------------------------------------
__global__ void k_scores_gemm(const f16* __restrict__ Qh,
                              const f16* __restrict__ Kh,
                              f16* __restrict__ S) {
  const int lane    = threadIdx.x & 31;
  const int wave    = threadIdx.x >> 5;
  const int halfsel = lane >> 4;
  const int lr      = lane & 15;
  const int t       = blockIdx.x * 8 + wave;    // 0 .. 4095
  const int b       = blockIdx.y;
  const int qt      = t >> 5;                   // 0 .. 127
  const int ntg     = t & 31;                   // nt group of 4
  const int nt0     = ntg * 4;
  if (nt0 > qt) return;                         // wave-uniform exit

  const f16* Qb = Qh + (size_t)b * ATT_N * ATT_D;
  const f16* Kb = Kh + (size_t)b * ATT_N * ATT_D;
  const f16* arow  = Qb + (size_t)(qt * 16 + lr) * ATT_D + halfsel * 16;
  const f16* brow0 = Kb + (size_t)((nt0 + 0) * 16 + lr) * ATT_D + halfsel * 16;
  const f16* brow1 = Kb + (size_t)((nt0 + 1) * 16 + lr) * ATT_D + halfsel * 16;
  const f16* brow2 = Kb + (size_t)((nt0 + 2) * 16 + lr) * ATT_D + halfsel * 16;
  const f16* brow3 = Kb + (size_t)((nt0 + 3) * 16 + lr) * ATT_D + halfsel * 16;

  v8f acc0 = {}, acc1 = {}, acc2 = {}, acc3 = {};
  for (int k0 = 0; k0 < ATT_D; k0 += 32) {
    v16h a = make_afrag(arow + k0);
    acc0 = wmma_f16(a, *(const v16h*)(brow0 + k0), acc0);
    acc1 = wmma_f16(a, *(const v16h*)(brow1 + k0), acc1);
    acc2 = wmma_f16(a, *(const v16h*)(brow2 + k0), acc2);
    acc3 = wmma_f16(a, *(const v16h*)(brow3 + k0), acc3);
  }

  const float scale = 0.03125f;                 // 1/sqrt(1024)
  for (int j = 0; j < 4; ++j) {
    const int nt = nt0 + j;
    if (nt > qt) break;                         // store only causal tiles
    v8f acc = (j == 0) ? acc0 : (j == 1) ? acc1 : (j == 2) ? acc2 : acc3;
    const int n = nt * 16 + lr;
    for (int r = 0; r < 8; ++r) {
      int q = qt * 16 + halfsel * 8 + r;
      float val = acc[r] * scale;
      if (n > q) val = NEG_INF;                 // mask inside diagonal tile
      S[((size_t)b * ATT_N + q) * ATT_N + n] = (f16)val;
    }
  }
}

// ---------------------------------------------------------------------------
// Kernel 4: in-place row softmax over padded causal length; zero-fills the
// rest of the row so P@V consumes clean 32-wide K chunks.  One block per row.
// ---------------------------------------------------------------------------
__global__ void k_softmax_rows(f16* __restrict__ S) {
  const int row = blockIdx.x;                   // 0 .. B*N-1
  const int q   = row & (ATT_N - 1);
  f16* Srow = S + (size_t)row * ATT_N;
  const int L = ((q >> 4) + 1) << 4;            // causal length padded to 16

  __shared__ float red[256];
  const int tid = threadIdx.x;

  float lmax = NEG_INF;
  for (int i = tid; i < L; i += 256) lmax = fmaxf(lmax, (float)Srow[i]);
  red[tid] = lmax; __syncthreads();
  for (int s = 128; s > 0; s >>= 1) {
    if (tid < s) red[tid] = fmaxf(red[tid], red[tid + s]);
    __syncthreads();
  }
  const float m = red[0]; __syncthreads();

  float lsum = 0.0f;
  for (int i = tid; i < L; i += 256) {
    float e = __expf((float)Srow[i] - m);       // exp(-inf - m) == 0
    lsum += e;
    Srow[i] = (f16)e;
  }
  red[tid] = lsum; __syncthreads();
  for (int s = 128; s > 0; s >>= 1) {
    if (tid < s) red[tid] += red[tid + s];
    __syncthreads();
  }
  const float inv = 1.0f / red[0];

  for (int i = tid; i < L; i += 256) Srow[i] = (f16)((float)Srow[i] * inv);
  for (int i = L + tid; i < ATT_N; i += 256) Srow[i] = (f16)0.0f;  // zero pad
}

// ---------------------------------------------------------------------------
// Kernel 5: context = P @ V.  V stored d-major so B-fragments are contiguous
// Vt-row chunks.  Each wave: 16 q-rows x 64 d-cols (4 WMMAs reuse the P
// fragment).  K loop bounded by the causal extent of the query tile.
// ---------------------------------------------------------------------------
__global__ void k_pv_gemm(const f16* __restrict__ P,
                          const f16* __restrict__ Vt,
                          float* __restrict__ out) {
  const int lane    = threadIdx.x & 31;
  const int wave    = threadIdx.x >> 5;
  const int halfsel = lane >> 4;
  const int lr      = lane & 15;
  const int qt = blockIdx.x;                     // 0..127
  const int b  = blockIdx.z;                     // 0..7
  const int d0 = (blockIdx.y * 8 + wave) * 64;   // grid.y = 2 -> covers D

  const f16* Prow  = P  + ((size_t)b * ATT_N + qt * 16 + lr) * ATT_N + halfsel * 16;
  const f16* vbase = Vt + ((size_t)b * ATT_D + lr) * ATT_N + halfsel * 16;
  const f16* vrow0 = vbase + (size_t)(d0 + 0)  * ATT_N;
  const f16* vrow1 = vbase + (size_t)(d0 + 16) * ATT_N;
  const f16* vrow2 = vbase + (size_t)(d0 + 32) * ATT_N;
  const f16* vrow3 = vbase + (size_t)(d0 + 48) * ATT_N;
  const int ksteps = ((qt + 1) * 16 + 31) >> 5;  // ceil(causal_len / 32)

  v8f acc0 = {}, acc1 = {}, acc2 = {}, acc3 = {};
  for (int ks = 0; ks < ksteps; ++ks) {
    int k0 = ks * 32;
    v16h a = make_afrag(Prow + k0);
    acc0 = wmma_f16(a, *(const v16h*)(vrow0 + k0), acc0);
    acc1 = wmma_f16(a, *(const v16h*)(vrow1 + k0), acc1);
    acc2 = wmma_f16(a, *(const v16h*)(vrow2 + k0), acc2);
    acc3 = wmma_f16(a, *(const v16h*)(vrow3 + k0), acc3);
  }

  for (int r = 0; r < 8; ++r) {
    size_t row = ((size_t)b * ATT_N + qt * 16 + halfsel * 8 + r) * ATT_D;
    out[row + d0 + 0  + lr] = acc0[r];
    out[row + d0 + 16 + lr] = acc1[r];
    out[row + d0 + 32 + lr] = acc2[r];
    out[row + d0 + 48 + lr] = acc3[r];
  }
}

// ---------------------------------------------------------------------------
// Host-side launch.  Workspace layout (f16 elements), ~208 MB total:
//   Xh [M*D] | Wh [3*D*D] | Qh [M*D] | Kh [M*D] | Vt [M*D] | S [B*N*N]
// ---------------------------------------------------------------------------
extern "C" void kernel_launch(void* const* d_in, const int* in_sizes, int n_in,
                              void* d_out, int out_size, void* d_ws, size_t ws_size,
                              hipStream_t stream) {
  const float* x  = (const float*)d_in[0];
  const float* Wq = (const float*)d_in[1];
  const float* Wk = (const float*)d_in[2];
  const float* Wv = (const float*)d_in[3];

  f16* Xh = (f16*)d_ws;
  f16* Wh = Xh + (size_t)ATT_M * ATT_D;
  f16* Qh = Wh + (size_t)3 * ATT_D * ATT_D;
  f16* Kh = Qh + (size_t)ATT_M * ATT_D;
  f16* Vt = Kh + (size_t)ATT_M * ATT_D;
  f16* S  = Vt + (size_t)ATT_M * ATT_D;

  // 1) casts (vectorized x4)
  {
    int n4 = (ATT_M * ATT_D) / 4;
    k_cast_f32_f16<<<dim3(n4 / 256), dim3(256), 0, stream>>>(x, Xh, n4);
    int w4 = (ATT_D * ATT_D) / 4;
    k_cast_f32_f16<<<dim3(w4 / 256), dim3(256), 0, stream>>>(Wq, Wh, w4);
    k_cast_f32_f16<<<dim3(w4 / 256), dim3(256), 0, stream>>>(Wk, Wh + (size_t)ATT_D * ATT_D, w4);
    k_cast_f32_f16<<<dim3(w4 / 256), dim3(256), 0, stream>>>(Wv, Wh + (size_t)2 * ATT_D * ATT_D, w4);
  }

  // 2) Q/K/V projections (z: 0=Q,1=K,2=V-transposed), async-LDS B staging
  k_qkv_gemm<<<dim3(ATT_M / 128, ATT_D / 64, 3), dim3(256), 0, stream>>>(Xh, Wh, Qh, Vt);

  // 3) causal scores: 128 qt x 32 nt-groups x 8 batches, 8 waves/block
  k_scores_gemm<<<dim3(512, ATT_B), dim3(256), 0, stream>>>(Qh, Kh, S);

  // 4) row softmax (in place, zero-pads tail)
  k_softmax_rows<<<dim3(ATT_B * ATT_N), dim3(256), 0, stream>>>(S);

  // 5) context = P @ V -> f32 out
  k_pv_gemm<<<dim3(ATT_N / 16, 2, ATT_B), dim3(256), 0, stream>>>(S, Vt, (float*)d_out);
}
